// KoopmanOperators_17205638988031
// MI455X (gfx1250) — matrix operand order repaired and checked by reading, not compiled, not run on hardware
//
#include <hip/hip_runtime.h>
#include <hip/hip_bf16.h>

// ---------------------------------------------------------------------------
// KoopmanOperators relation-network forward, MI455X (gfx1250, wave32, WMMA).
// All GEMMs on v_wmma_f32_16x16x32_bf16 (fp32 accumulate).  Weights are
// pre-swizzled into WMMA fragment-linear layout so every LDS fragment load is
// ds_load_b128; the pairwise GEMM1->GEMM2 handoff is a pure register
// re-interpretation (transposed GEMM1), no LDS staging.
// ---------------------------------------------------------------------------

typedef __attribute__((ext_vector_type(16))) __bf16 v16bf;
typedef __attribute__((ext_vector_type(8)))  __bf16 v8bf;
typedef __attribute__((ext_vector_type(8)))  float  v8f;
typedef __attribute__((ext_vector_type(4)))  float  f4;

static __device__ __forceinline__ v8f wmma_bf16(v16bf a, v16bf b, v8f c) {
  return __builtin_amdgcn_wmma_f32_16x16x32_bf16(false, a, false, b, (short)0, c,
                                                 false, false);
}
static __device__ __forceinline__ v16bf ldfrag(const __bf16* p) {
  union { v16bf v; v8bf h[2]; } u;
  u.h[0] = *(const v8bf*)(p);
  u.h[1] = *(const v8bf*)(p + 8);
  return u.v;
}
static __device__ __forceinline__ v16bf ldfrag2(const __bf16* p0, const __bf16* p1) {
  union { v16bf v; v8bf h[2]; } u;
  u.h[0] = *(const v8bf*)(p0);
  u.h[1] = *(const v8bf*)(p1);
  return u.v;
}

// problem sizes
constexpr int NODES = 8192;            // BT(128) * N(64)

// f32 workspace layout (element offsets)
constexpr int OFF_P    = 0;                       // [NODES][128]  s@reW1 (no bias)
constexpr int OFF_EOBJ = OFF_P    + NODES * 128;  // [NODES][128]  enc_obj
constexpr int OFF_OBSU = OFF_EOBJ + NODES * 128;  // [NODES][32]   obs_u
constexpr int OFF_R    = OFF_OBSU + NODES * 32;   // [NODES][64]   enc_obj@rpW[0:128]
constexpr int OFF_S    = OFF_R    + NODES * 64;   // [NODES][64]   enc_obj@rpW[128:256]
constexpr int OFF_AGG  = OFF_S    + NODES * 64;   // [NODES][64]   rel_agg
constexpr int OFF_SX   = OFF_AGG  + NODES * 64;   // [NODES]
constexpr int OFF_SY   = OFF_SX   + NODES;        // [NODES]
constexpr int F32_TOTAL = OFF_SY + NODES;
constexpr size_t BF16_BYTE = (size_t)F32_TOTAL * 4;

// bf16 (pre-swizzled weights) layout, element offsets inside bf16 region
constexpr int OFF_W1S  = 0;                 // reW2^T  A-layout: 4mt x 4ks x 512 = 8192
constexpr int OFF_WBS  = OFF_W1S  + 8192;   // rpW_bot B-layout: 2ks x 4nt x 512 = 4096
constexpr int OFF_CW1S = OFF_WBS  + 4096;   // cW1^T   A-layout: 8mt x 7ks x 512 = 28672
constexpr int OFF_CW2S = OFF_CW1S + 28672;  // cW2     B-layout: 4ks x 8nt x 512 = 16384
constexpr int OFF_CW3S = OFF_CW2S + 16384;  // cW3     B-layout: 4ks x 2nt x 512 = 4096

// dynamic LDS sizes
constexpr int PAIR_LDS = (64*128 + 64*72 + 16*64 + 16*64 + 64 + 64) * 4
                       + (64*128 + 64*64) * 2;                  // 84480 B
constexpr int HEAD_LDS = (28672 + 16384 + 4096 + 4*16*224) * 2; // 126976 B

// ---------------------------------------------------------------------------
// Weight swizzlers: emit tile-linear fragments so a lane's 16 bf16 elements
// are contiguous (2 x ds_load_b128 per fragment inside the GEMM kernels).
// A-layout (matrix used as WMMA A, source read TRANSPOSED: A[m][k]=src[k][m]):
//   dst[((mt*KS+ks)*32 + lane)*16 + e] , K = ks*32 + hi*16 + e
// B-layout (matrix used as WMMA B, B[k][n]=src[k][n]):
//   dst[((ks*NT+nt)*32 + lane)*16 + e] , K = ks*32 + hi*8 + (e<8 ? e : e+8)
// ---------------------------------------------------------------------------
__global__ void swizzleA_kernel(const float* __restrict__ src, __bf16* __restrict__ dst,
                                int KS, int srcCols, int total) {
  int o = blockIdx.x * 256 + threadIdx.x;
  if (o >= total) return;
  int e = o & 15, l = (o >> 4) & 31, t = o >> 9;
  int ks = t % KS, mt = t / KS;
  int m = mt * 16 + (l & 15);
  int k = ks * 32 + (l >> 4) * 16 + e;
  dst[o] = (__bf16)src[k * srcCols + m];
}

__global__ void swizzleB_kernel(const float* __restrict__ src, __bf16* __restrict__ dst,
                                int NT, int N, int total) {
  int o = blockIdx.x * 256 + threadIdx.x;
  if (o >= total) return;
  int e = o & 15, l = (o >> 4) & 31, t = o >> 9;
  int nt = t % NT, ks = t / NT;
  int n = nt * 16 + (l & 15);
  int k = ks * 32 + (l >> 4) * 8 + (e < 8 ? e : e + 8);
  dst[o] = (__bf16)src[k * N + n];
}

// ---------------------------------------------------------------------------
// Per-node phase: enc_obj, obs_u, P = s@reW1, R, S, sx/sy.  One block per node.
// (<5% of total FLOPs; plain VALU.)
// ---------------------------------------------------------------------------
__global__ void node_kernel(const float* __restrict__ states,
    const float* __restrict__ sW1, const float* __restrict__ sb1,
    const float* __restrict__ sW2, const float* __restrict__ sb2,
    const float* __restrict__ uW1, const float* __restrict__ ub1,
    const float* __restrict__ uW2, const float* __restrict__ ub2,
    const float* __restrict__ upW, const float* __restrict__ upb,
    const float* __restrict__ reW1, const float* __restrict__ rpW,
    float* __restrict__ wf) {
  __shared__ float sdat[8], h1s[128], hu[128], eobj[128], eu[32];
  const int t = threadIdx.x;
  const int node = blockIdx.x;
  const int btI = node >> 6, n = node & 63;
  const int b = btI >> 3, tt = btI & 7;        // bt = b*T + t  (T=8)
  if (t < 8) sdat[t] = states[(((size_t)(b * 64 + n)) * 8 + tt) * 8 + t];
  __syncthreads();

  float a1 = sb1[t], au = ub1[t], ap = 0.f;
#pragma unroll
  for (int k = 0; k < 8; ++k) {
    float sv = sdat[k];
    a1 += sv * sW1[k * 128 + t];
    au += sv * uW1[k * 128 + t];
    ap += sv * reW1[k * 128 + t];
  }
  h1s[t] = fmaxf(a1, 0.f);
  hu[t]  = fmaxf(au, 0.f);
  wf[OFF_P + (size_t)node * 128 + t] = ap;
  if (t == 0) { wf[OFF_SX + node] = sdat[0]; wf[OFF_SY + node] = sdat[4]; }
  __syncthreads();

  float e2 = sb2[t];
  for (int k = 0; k < 128; ++k) e2 += h1s[k] * sW2[k * 128 + t];
  e2 = fmaxf(e2, 0.f);
  eobj[t] = e2;
  wf[OFF_EOBJ + (size_t)node * 128 + t] = e2;
  if (t < 32) {
    float v = ub2[t];
    for (int k = 0; k < 128; ++k) v += hu[k] * uW2[k * 32 + t];
    eu[t] = v;                                  // enc_u (no relu in this path)
  }
  __syncthreads();

  if (t < 32) {
    float v = upb[t];
    for (int k = 0; k < 128; ++k) v += eobj[k] * upW[k * 32 + t];
    for (int k = 0; k < 32; ++k)  v += eu[k] * upW[(128 + k) * 32 + t];
    wf[OFF_OBSU + (size_t)node * 32 + t] = fmaxf(v, 0.f);
  }
  if (t < 64) {
    float r = 0.f;
    for (int k = 0; k < 128; ++k) r += eobj[k] * rpW[k * 64 + t];
    wf[OFF_R + (size_t)node * 64 + t] = r;
  } else {
    const int cc = t - 64;
    float sv = 0.f;
    for (int k = 0; k < 128; ++k) sv += eobj[k] * rpW[(128 + k) * 64 + cc];
    wf[OFF_S + (size_t)node * 64 + cc] = sv;
  }
}

// ---------------------------------------------------------------------------
// Pairwise phase.  Grid: BT(128) x i-tiles(4).  Block: 4 waves; wave w owns
// senders [16w,16w+16).  Per receiver i:
//   GEMM1 (transposed): enc_rel^T = (reW2^T)_A x (h1^T)_B  -> D tiles land in
//   registers already in the A-layout GEMM2 wants.
//   GEMM2 (normal):     rel = enc_rel_A x (rpW_bot)_B, + R_i + S_j + b,
//   relu*sel, reduce over senders (v-sum + one shfl_xor(16)), ds_add_f32.
// ---------------------------------------------------------------------------
__global__ void pair_kernel(float* __restrict__ wf, const __bf16* __restrict__ wb,
                            const float* __restrict__ reb1,
                            const float* __restrict__ reb2,
                            const float* __restrict__ rpb) {
  extern __shared__ char smem[];
  float* Pl  = (float*)smem;            // [64][128]
  float* SlT = Pl + 64 * 128;           // S transposed, [64 cols][72 pad]
  float* Rl  = SlT + 64 * 72;           // [16][64]
  float* agg = Rl + 16 * 64;            // [16][64]
  float* sxl = agg + 16 * 64;           // [64]
  float* syl = sxl + 64;                // [64]
  __bf16* W1s = (__bf16*)(syl + 64);    // reW2^T A-swz, 16 tiles x 512
  __bf16* Wbs = W1s + 64 * 128;         // rpW_bot B-swz, 8 tiles x 512

  const int tid = threadIdx.x, lane = tid & 31, w = tid >> 5;
  const int bt = blockIdx.x >> 2, itile = blockIdx.x & 3;
  const int nb = bt * 64;

  const float* Pg = wf + OFF_P + (size_t)nb * 128;
  const float* Sg = wf + OFF_S + (size_t)nb * 64;
  const float* Rg = wf + OFF_R + (size_t)(nb + itile * 16) * 64;
  for (int idx = tid; idx < 64 * 128; idx += 128) Pl[idx] = Pg[idx];
  for (int idx = tid; idx < 64 * 64;  idx += 128) SlT[(idx & 63) * 72 + (idx >> 6)] = Sg[idx];
  for (int idx = tid; idx < 16 * 64;  idx += 128) { Rl[idx] = Rg[idx]; agg[idx] = 0.f; }
  for (int idx = tid; idx < 64 * 128; idx += 128) W1s[idx] = wb[OFF_W1S + idx];
  for (int idx = tid; idx < 64 * 64;  idx += 128) Wbs[idx] = wb[OFF_WBS + idx];
  if (tid < 64) { sxl[tid] = wf[OFF_SX + nb + tid]; syl[tid] = wf[OFF_SY + nb + tid]; }
  __syncthreads();

  const int cpos = lane & 15, hi = lane >> 4;
  const int jbase = w * 16;
  const int pj = jbase + cpos;                 // this lane's sender column
  const f4* PjV = (const f4*)(Pl + pj * 128);
  const f4* SlTV = (const f4*)SlT;
  const f4* rb1V = (const f4*)reb1;

  // loop-invariant per-lane data
  f4 rb1v[16];
#pragma unroll
  for (int ks = 0; ks < 4; ++ks)
#pragma unroll
    for (int q = 0; q < 4; ++q)
      rb1v[ks * 4 + q] = rb1V[ks * 8 + hi * 4 + q];

  float rb2lo[16], rb2hi[16];
#pragma unroll
  for (int ks2 = 0; ks2 < 2; ++ks2)
#pragma unroll
    for (int e = 0; e < 8; ++e) {
      rb2lo[ks2 * 8 + e] = reb2[ks2 * 32 + hi * 8 + e];
      rb2hi[ks2 * 8 + e] = reb2[ks2 * 32 + 16 + hi * 8 + e];
    }
  float sxj[8], syj[8];
#pragma unroll
  for (int v = 0; v < 8; ++v) {
    sxj[v] = sxl[jbase + hi * 8 + v];
    syj[v] = syl[jbase + hi * 8 + v];
  }
  float rpbv[4];
#pragma unroll
  for (int nt = 0; nt < 4; ++nt) rpbv[nt] = rpb[nt * 16 + cpos];

  for (int i = 0; i < 16; ++i) {
    const int il = itile * 16 + i;
    const f4* PiV = (const f4*)(Pl + il * 128);
    // B1 = h1^T = relu(P_i - P_j + reb1)^T  (vector LDS reads, lane = sender)
    v16bf b1[4];
#pragma unroll
    for (int ks = 0; ks < 4; ++ks) {
#pragma unroll
      for (int q = 0; q < 4; ++q) {
        const int fi = ks * 8 + hi * 4 + q;
        const f4 pi = PiV[fi], pjv = PjV[fi], rb = rb1v[ks * 4 + q];
#pragma unroll
        for (int z = 0; z < 4; ++z)
          b1[ks][q * 4 + z] = (__bf16)fmaxf(pi[z] - pjv[z] + rb[z], 0.f);
      }
    }
    // GEMM1': enc_rel^T tiles (pre bias/relu)
    v8f d1[4];
#pragma unroll
    for (int mt = 0; mt < 4; ++mt) {
      v8f acc = {0.f,0.f,0.f,0.f,0.f,0.f,0.f,0.f};
#pragma unroll
      for (int ks = 0; ks < 4; ++ks) {
        const v16bf a1 = ldfrag(W1s + ((mt * 4 + ks) * 32 + lane) * 16);
        acc = wmma_bf16(a1, b1[ks], acc);
      }
      d1[mt] = acc;
    }
    // A2 = enc_rel rows: pure register re-interpretation of the D tiles
    v16bf a2f[2];
#pragma unroll
    for (int ks2 = 0; ks2 < 2; ++ks2)
#pragma unroll
      for (int e = 0; e < 8; ++e) {
        a2f[ks2][e]     = (__bf16)fmaxf(d1[ks2 * 2][e]     + rb2lo[ks2 * 8 + e], 0.f);
        a2f[ks2][e + 8] = (__bf16)fmaxf(d1[ks2 * 2 + 1][e] + rb2hi[ks2 * 8 + e], 0.f);
      }
    // sel mask per sender row (shared across all 4 column tiles)
    const float sxi = sxl[il], syi = syl[il];
    float selm[8];
#pragma unroll
    for (int v = 0; v < 8; ++v)
      selm[v] = (fabsf(sxi - sxj[v]) > 0.1f || fabsf(syi - syj[v]) > 0.1f) ? 1.f : 0.f;
    // GEMM2 + epilogue
#pragma unroll
    for (int nt = 0; nt < 4; ++nt) {
      v8f acc = {0.f,0.f,0.f,0.f,0.f,0.f,0.f,0.f};
#pragma unroll
      for (int ks2 = 0; ks2 < 2; ++ks2) {
        const v16bf bw = ldfrag(Wbs + ((ks2 * 4 + nt) * 32 + lane) * 16);
        acc = wmma_bf16(a2f[ks2], bw, acc);
      }
      const int cc = nt * 16 + cpos;
      const float base = Rl[i * 64 + cc] + rpbv[nt];
      const int sIdx = cc * 18 + (jbase >> 2) + hi * 2;   // f4 units
      const f4 s0 = SlTV[sIdx], s1 = SlTV[sIdx + 1];
      float sum = 0.f;
#pragma unroll
      for (int v = 0; v < 8; ++v) {
        const float sv = (v < 4) ? s0[v] : s1[v - 4];
        sum += fmaxf(acc[v] + base + sv, 0.f) * selm[v];
      }
      sum += __shfl_xor(sum, 16, 32);   // combine the two lane halves
      if (lane < 16) atomicAdd(&agg[i * 64 + cc], sum);
    }
  }
  __syncthreads();
  float* aggG = wf + OFF_AGG + (size_t)(nb + itile * 16) * 64;
  for (int idx = tid; idx < 16 * 64; idx += 128) aggG[idx] = agg[idx];
}

// ---------------------------------------------------------------------------
// Head MLP: [16,224] -> 128 relu -> 128 relu -> 32.  4 waves/block, each wave
// owns a 16-node group.  Layer1 runs transposed so its D tiles feed layer2 as
// A-fragments straight from registers; only layer2->layer3 goes through LDS.
// ---------------------------------------------------------------------------
__global__ void head_kernel(const float* __restrict__ wf,
                            const __bf16* __restrict__ wb,
                            const float* __restrict__ cb1,
                            const float* __restrict__ cb2,
                            const float* __restrict__ cb3,
                            float* __restrict__ out) {
  extern __shared__ char smem[];
  __bf16* W1s = (__bf16*)smem;          // cW1^T A-swz: 56 tiles x 512
  __bf16* W2s = W1s + 28672;            // cW2 B-swz: 32 tiles x 512
  __bf16* W3s = W2s + 16384;            // cW3 B-swz: 8 tiles x 512
  __bf16* Ast = W3s + 4096;             // 4 waves x [16][224]

  const int tid = threadIdx.x, lane = tid & 31, w = tid >> 5;
  for (int idx = tid; idx < 28672; idx += 128) W1s[idx] = wb[OFF_CW1S + idx];
  for (int idx = tid; idx < 16384; idx += 128) W2s[idx] = wb[OFF_CW2S + idx];
  for (int idx = tid; idx < 4096;  idx += 128) W3s[idx] = wb[OFF_CW3S + idx];
  __syncthreads();

  const int g = blockIdx.x * 4 + w;      // 16-node group
  __bf16* A = Ast + w * (16 * 224);
  for (int idx = lane; idx < 16 * 224; idx += 32) {
    const int row = idx / 224, k = idx - row * 224;
    const int node = g * 16 + row;
    float v;
    if (k < 128)      v = wf[OFF_EOBJ + (size_t)node * 128 + k];
    else if (k < 160) v = wf[OFF_OBSU + (size_t)node * 32 + (k - 128)];
    else              v = wf[OFF_AGG  + (size_t)node * 64 + (k - 160)];
    A[idx] = (__bf16)v;
  }
  const int cpos = lane & 15, hi = lane >> 4;

  // B-fragments of input^T (lane = node column); vector LDS reads
  v16bf bin[7];
#pragma unroll
  for (int ks = 0; ks < 7; ++ks)
    bin[ks] = ldfrag(A + cpos * 224 + ks * 32 + hi * 16);

  // layer1 (transposed): d1[mt] = (x @ cW1)^T tiles
  v8f d1[8];
#pragma unroll
  for (int mt = 0; mt < 8; ++mt) {
    v8f acc = {0.f,0.f,0.f,0.f,0.f,0.f,0.f,0.f};
#pragma unroll
    for (int ks = 0; ks < 7; ++ks) {
      const v16bf a1 = ldfrag(W1s + ((mt * 7 + ks) * 32 + lane) * 16);
      acc = wmma_bf16(a1, bin[ks], acc);
    }
    d1[mt] = acc;
  }
  // layer2 A-fragments straight from layer1 D registers (+bias, relu)
  v16bf a2f[4];
#pragma unroll
  for (int ks = 0; ks < 4; ++ks)
#pragma unroll
    for (int e = 0; e < 8; ++e) {
      const int kA = ks * 32 + hi * 8 + e, kB = kA + 16;
      a2f[ks][e]     = (__bf16)fmaxf(d1[ks * 2][e]     + cb1[kA], 0.f);
      a2f[ks][e + 8] = (__bf16)fmaxf(d1[ks * 2 + 1][e] + cb1[kB], 0.f);
    }
  // layer2 (normal) -> stage h2 rows into LDS (reuse Ast)
#pragma unroll
  for (int nt = 0; nt < 8; ++nt) {
    v8f acc = {0.f,0.f,0.f,0.f,0.f,0.f,0.f,0.f};
#pragma unroll
    for (int ks = 0; ks < 4; ++ks) {
      const v16bf bw = ldfrag(W2s + ((ks * 8 + nt) * 32 + lane) * 16);
      acc = wmma_bf16(a2f[ks], bw, acc);
    }
    const int c2 = nt * 16 + cpos;
#pragma unroll
    for (int v = 0; v < 8; ++v)
      A[(v + hi * 8) * 224 + c2] = (__bf16)fmaxf(acc[v] + cb2[c2], 0.f);
  }
  // layer3 (normal): A-fragments from stage (vector reads), write output
  v16bf a3f[4];
#pragma unroll
  for (int ks = 0; ks < 4; ++ks)
    a3f[ks] = ldfrag2(A + cpos * 224 + ks * 32 + hi * 8,
                      A + cpos * 224 + ks * 32 + 16 + hi * 8);
#pragma unroll
  for (int nt = 0; nt < 2; ++nt) {
    v8f acc = {0.f,0.f,0.f,0.f,0.f,0.f,0.f,0.f};
#pragma unroll
    for (int ks = 0; ks < 4; ++ks) {
      const v16bf bw = ldfrag(W3s + ((ks * 2 + nt) * 32 + lane) * 16);
      acc = wmma_bf16(a3f[ks], bw, acc);
    }
    const int c3 = nt * 16 + cpos;
#pragma unroll
    for (int v = 0; v < 8; ++v) {
      const int node = g * 16 + v + hi * 8;
      out[(size_t)node * 32 + c3] = acc[v] + cb3[c3];
    }
  }
}

// ---------------------------------------------------------------------------
extern "C" void kernel_launch(void* const* d_in, const int* in_sizes, int n_in,
                              void* d_out, int out_size, void* d_ws, size_t ws_size,
                              hipStream_t stream) {
  const float* states = (const float*)d_in[0];
  // d_in[1] = temp (unused by the reference math)
  const float* sW1  = (const float*)d_in[2];  const float* sb1 = (const float*)d_in[3];
  const float* sW2  = (const float*)d_in[4];  const float* sb2 = (const float*)d_in[5];
  const float* uW1  = (const float*)d_in[6];  const float* ub1 = (const float*)d_in[7];
  const float* uW2  = (const float*)d_in[8];  const float* ub2 = (const float*)d_in[9];
  const float* upW  = (const float*)d_in[10]; const float* upb = (const float*)d_in[11];
  const float* reW1 = (const float*)d_in[12]; const float* reb1= (const float*)d_in[13];
  const float* reW2 = (const float*)d_in[14]; const float* reb2= (const float*)d_in[15];
  const float* rpW  = (const float*)d_in[16]; const float* rpb = (const float*)d_in[17];
  const float* cW1  = (const float*)d_in[18]; const float* cb1 = (const float*)d_in[19];
  const float* cW2  = (const float*)d_in[20]; const float* cb2 = (const float*)d_in[21];
  const float* cW3  = (const float*)d_in[22]; const float* cb3 = (const float*)d_in[23];

  float*  wf = (float*)d_ws;
  __bf16* wb = (__bf16*)((char*)d_ws + BF16_BYTE);
  float*  out = (float*)d_out;

  // pre-swizzle weights into fragment-linear bf16 (tiny, one-off)
  swizzleA_kernel<<<(8192  + 255) / 256, 256, 0, stream>>>(reW2,          wb + OFF_W1S,  4, 64,  8192);
  swizzleB_kernel<<<(4096  + 255) / 256, 256, 0, stream>>>(rpW + 256*64,  wb + OFF_WBS,  4, 64,  4096);
  swizzleA_kernel<<<(28672 + 255) / 256, 256, 0, stream>>>(cW1,           wb + OFF_CW1S, 7, 128, 28672);
  swizzleB_kernel<<<(16384 + 255) / 256, 256, 0, stream>>>(cW2,           wb + OFF_CW2S, 8, 128, 16384);
  swizzleB_kernel<<<(4096  + 255) / 256, 256, 0, stream>>>(cW3,           wb + OFF_CW3S, 2, 32,  4096);

  node_kernel<<<NODES, 128, 0, stream>>>(states, sW1, sb1, sW2, sb2, uW1, ub1,
                                         uW2, ub2, upW, upb, reW1, rpW, wf);
  pair_kernel<<<512, 128, PAIR_LDS, stream>>>(wf, wb, reb1, reb2, rpb);
  head_kernel<<<128, 128, HEAD_LDS, stream>>>(wf, wb, cb1, cb2, cb3, out);
}
